// gconv_residual_50208167690604
// MI455X (gfx1250) — compile-verified, matrix operand order checked by reading, not run on hardware
//
#include <hip/hip_runtime.h>
#include <hip/hip_bf16.h>
#include <math.h>

// Shapes
#define BB    4
#define NIN   64
#define NMED  64
#define NET   4
#define NN    2048
#define KK    32
#define MM    (BB * NN)        // 8192 rows (b,n)
#define EPSV  1e-5f

typedef __attribute__((ext_vector_type(16))) _Float16 v16h;
typedef __attribute__((ext_vector_type(8)))  _Float16 v8h;
typedef __attribute__((ext_vector_type(8)))  float    v8f;

// ---------------------------------------------------------------------------
// prep: node_feature [B,64,N] -> Xh f16 [M=8192, 64] row-major
// ---------------------------------------------------------------------------
__global__ void prep_x(const float* __restrict__ nf_in, _Float16* __restrict__ Xh) {
  int i = blockIdx.x * blockDim.x + threadIdx.x;   // 524288 = m*64 + c
  int c = i & 63;
  int m = i >> 6;
  int b = m >> 11;
  int n = m & 2047;
  Xh[i] = (_Float16)nf_in[(((size_t)b * 64 + c) << 11) + n];
}

// ---------------------------------------------------------------------------
// prep weights: W1h [o][c], W2h [i][o] (both already [out,in] row-major),
// FcatT [512][64] f16 column-pretransposed combined filters:
//   n<256: F_top[k][n]+F_bot[k][n]  (for P)    n>=256: F_bot[k][n-256] (for Q)
// ---------------------------------------------------------------------------
__global__ void prep_w(const float* __restrict__ w1, const float* __restrict__ w2,
                       const float* __restrict__ filters,
                       _Float16* __restrict__ W1h, _Float16* __restrict__ W2h,
                       _Float16* __restrict__ FcatT) {
  int i = blockIdx.x * blockDim.x + threadIdx.x;   // 40960 total
  if (i < 4096) {
    W1h[i] = (_Float16)w1[i];
  } else if (i < 8192) {
    W2h[i - 4096] = (_Float16)w2[i - 4096];
  } else if (i < 40960) {
    int j = i - 8192;
    int nn = j >> 6, k = j & 63;
    float v = (nn < 256)
                  ? (filters[k * 256 + nn] + filters[(k + 64) * 256 + nn])
                  : filters[(k + 64) * 256 + (nn - 256)];
    FcatT[j] = (_Float16)v;
  }
}

// ---------------------------------------------------------------------------
// WMMA fragment helpers (CDNA5 ISA 7.12.2, 16-bit A 16x32 layout):
//   lanes 0-15 (row m)  : halves 0..7 = K+0..7,  halves 8..15 = K+16..23
//   lanes 16-31 (row m) : halves 0..7 = K+8..15, halves 8..15 = K+24..31
// Both A and B (pre-transposed to [cols][64]) load as two b128s per lane.
// ---------------------------------------------------------------------------
__device__ __forceinline__ v16h load_frag(const _Float16* p) {
  v8h lo = *(const v8h*)(p);
  v8h hi = *(const v8h*)(p + 16);
  v16h r;
#pragma unroll
  for (int q = 0; q < 8; ++q) { r[q] = lo[q]; r[q + 8] = hi[q]; }
  return r;
}

__device__ __forceinline__ v8f wmma_f16(v16h a, v16h b, v8f c) {
  return __builtin_amdgcn_wmma_f32_16x16x32_f16(false, a, false, b, (short)0, c,
                                                false, false);
}

// ---------------------------------------------------------------------------
// Generic WMMA GEMM: C[M, Ncols] = A[M,64] @ Bt[Ncols,64]^T   (f16 in, f32 out)
// Register-blocked 2x2: each wave computes a 32x32 output block = 4 tiles,
// K=64 -> 8x v_wmma_f32_16x16x32_f16, each A/B fragment reused twice.
// Requires M % 32 == 0 and Ncols % 32 == 0 (8192 and 64/512: OK).
// ---------------------------------------------------------------------------
__global__ __launch_bounds__(256) void gemm_wmma(
    const _Float16* __restrict__ A, const _Float16* __restrict__ Bt,
    float* __restrict__ C, int Ncols) {
  int wave = (blockIdx.x * blockDim.x + threadIdx.x) >> 5;
  int lane = threadIdx.x & 31;
  int nB2  = Ncols >> 5;               // # of 32-wide column blocks
  int mB   = wave / nB2;               // 32-row block
  int nB   = wave - mB * nB2;          // 32-col block
  int l15  = lane & 15;
  int kh   = (lane >> 4) << 3;         // 0 or 8

  const _Float16* arow0 = A  + (size_t)(mB * 32 + l15) * 64 + kh;
  const _Float16* arow1 = arow0 + 16 * 64;
  const _Float16* brow0 = Bt + (size_t)(nB * 32 + l15) * 64 + kh;
  const _Float16* brow1 = brow0 + 16 * 64;

  v8f acc00 = {}, acc01 = {}, acc10 = {}, acc11 = {};
#pragma unroll
  for (int ks = 0; ks < 64; ks += 32) {
    v16h a0 = load_frag(arow0 + ks);
    v16h a1 = load_frag(arow1 + ks);
    v16h b0 = load_frag(brow0 + ks);
    v16h b1 = load_frag(brow1 + ks);
    acc00 = wmma_f16(a0, b0, acc00);
    acc01 = wmma_f16(a0, b1, acc01);
    acc10 = wmma_f16(a1, b0, acc10);
    acc11 = wmma_f16(a1, b1, acc11);
  }
  // D layout: VGPR r -> M = r + (lane>=16 ? 8 : 0), N = lane&15
  int rb = mB * 32 + ((lane >> 4) << 3);
  int cb = nB * 32 + l15;
#pragma unroll
  for (int r = 0; r < 8; ++r) {
    C[(size_t)(rb + r) * Ncols + cb]           = acc00[r];
    C[(size_t)(rb + r) * Ncols + cb + 16]      = acc01[r];
    C[(size_t)(rb + 16 + r) * Ncols + cb]      = acc10[r];
    C[(size_t)(rb + 16 + r) * Ncols + cb + 16] = acc11[r];
  }
}

// ---------------------------------------------------------------------------
// Per-channel sums for train-mode BN (channel = col of [M,64] matrix)
// ---------------------------------------------------------------------------
__global__ void stats_kernel(const float* __restrict__ X,
                             float* __restrict__ sum, float* __restrict__ sumsq) {
  int t  = threadIdx.x;
  int c  = t & 63;
  int r0 = blockIdx.x * 4 + (t >> 6);   // gridDim = 64 -> 256 row-streams
  float s = 0.f, q = 0.f;
  for (int r = r0; r < MM; r += 256) {
    float v = X[(size_t)r * 64 + c];
    s += v; q += v * v;
  }
  atomicAdd(&sum[c], s);
  atomicAdd(&sumsq[c], q);
}

__global__ void finalize_bn(const float* __restrict__ sum, const float* __restrict__ sumsq,
                            const float* __restrict__ g, const float* __restrict__ b,
                            float* __restrict__ scale, float* __restrict__ shift) {
  int c = threadIdx.x;  // 64
  float mean = sum[c] * (1.f / (float)MM);
  float var  = sumsq[c] * (1.f / (float)MM) - mean * mean;
  float sc   = g[c] * rsqrtf(var + EPSV);
  scale[c] = sc;
  shift[c] = b[c] - mean * sc;
}

__global__ void bn_relu_to_f16(const float* __restrict__ X,
                               const float* __restrict__ scale,
                               const float* __restrict__ shift,
                               _Float16* __restrict__ Y) {
  int i = blockIdx.x * blockDim.x + threadIdx.x;  // 524288
  int c = i & 63;
  float v = fmaxf(X[i] * scale[c] + shift[c], 0.f);
  Y[i] = (_Float16)v;
}

// ---------------------------------------------------------------------------
// Edge mixing: ef[b,n,k,o] = sum_t (P[m, 4o+t] - Q[b*N+idx, 4o+t]) * et[b,t,n,k]
// agg[m,o] = max_k ef. PQ rows: cols 0..255 = P, 256..511 = Q. One block/node.
// Thread t -> (o = t>>2, tt = t&3); sum over NET=4 via in-wave shuffles.
// (mp_bias dropped: uniform per-channel shift cancels in training-mode BN.)
// ---------------------------------------------------------------------------
__global__ __launch_bounds__(256) void edge_kernel(
    const float* __restrict__ PQ, const int* __restrict__ nn_idx,
    const float* __restrict__ etype, float* __restrict__ aggraw) {
  int m = blockIdx.x;        // 0..8191
  int b = m >> 11, n = m & 2047;
  int t = threadIdx.x;
  int o = t >> 2, tt = t & 3;
  int f = (o << 2) | tt;

  float pv = PQ[(size_t)m * 512 + f];
  const int*   idxp = nn_idx + (size_t)m * KK;
  const float* etp  = etype + ((((size_t)b * NET + tt) * NN + n) * KK);

  float mx = -3.0e38f;
#pragma unroll 4
  for (int k = 0; k < KK; ++k) {
    int j = idxp[k];
    float qv = PQ[((size_t)((b << 11) + j)) * 512 + 256 + f];
    float v = (pv - qv) * etp[k];
    v += __shfl_xor(v, 1, 32);        // reduce over tt within groups of 4
    v += __shfl_xor(v, 2, 32);
    mx = fmaxf(mx, v);
  }
  if (tt == 0) aggraw[(size_t)m * 64 + o] = mx;
}

// ---------------------------------------------------------------------------
// Final: out[b,i,n] = relu(outraw[m,i]*scale+shift) + node_feature[b,i,n]
// ---------------------------------------------------------------------------
__global__ void final_kernel(const float* __restrict__ outraw,
                             const float* __restrict__ scale,
                             const float* __restrict__ shift,
                             const float* __restrict__ nf_in,
                             float* __restrict__ out) {
  int i = blockIdx.x * blockDim.x + threadIdx.x;  // 524288 in [b][c][n] order
  int b = i >> 17;
  int c = (i >> 11) & 63;
  int n = i & 2047;
  int m = (b << 11) | n;
  float v = outraw[(size_t)m * 64 + c] * scale[c] + shift[c];
  out[i] = fmaxf(v, 0.f) + nf_in[i];
}

// ---------------------------------------------------------------------------
extern "C" void kernel_launch(void* const* d_in, const int* in_sizes, int n_in,
                              void* d_out, int out_size, void* d_ws, size_t ws_size,
                              hipStream_t stream) {
  const float* nf_in   = (const float*)d_in[0];
  const float* etype   = (const float*)d_in[1];
  const float* conv1_w = (const float*)d_in[2];
  // d_in[3] conv1_b: cancels in train-mode BN
  const float* bn1_g   = (const float*)d_in[4];
  const float* bn1_b   = (const float*)d_in[5];
  const float* filters = (const float*)d_in[6];
  // d_in[7] mp_bias: cancels in train-mode BN
  const float* bnm_g   = (const float*)d_in[8];
  const float* bnm_b   = (const float*)d_in[9];
  const float* conv2_w = (const float*)d_in[10];
  // d_in[11] conv2_b: cancels in train-mode BN
  const float* bn2_g   = (const float*)d_in[12];
  const float* bn2_b   = (const float*)d_in[13];
  const int*   nn_idx  = (const int*)d_in[14];
  float* out = (float*)d_out;

  // Workspace layout (floats)
  float* ws      = (float*)d_ws;
  float* hraw    = ws;                       // 524288
  float* aggraw  = ws + 524288;              // 524288
  float* outraw  = ws + 1048576;             // 524288
  float* PQ      = ws + 1572864;             // 8192*512 = 4194304
  float* stats   = ws + 5767168;             // 3 sets * 256
  // per-set: +0 sum, +64 sumsq, +128 scale, +192 shift
  float* s0 = stats, * s1 = stats + 256, * s2 = stats + 512;
  _Float16* h16   = (_Float16*)(ws + 5767936);
  _Float16* Xh    = h16;                     // 524288
  _Float16* nfh   = Xh + 524288;             // 524288
  _Float16* aggh  = nfh + 524288;            // 524288
  _Float16* W1h   = aggh + 524288;           // 4096
  _Float16* W2h   = W1h + 4096;              // 4096
  _Float16* FcatT = W2h + 4096;              // 32768

  hipMemsetAsync(stats, 0, 768 * sizeof(float), stream);

  prep_x<<<2048, 256, 0, stream>>>(nf_in, Xh);
  prep_w<<<160, 256, 0, stream>>>(conv1_w, conv2_w, filters, W1h, W2h, FcatT);

  // conv1: hraw[8192,64] = Xh @ W1^T
  // waves = (8192/32)*(64/32) = 512 -> 64 blocks of 8 waves
  gemm_wmma<<<64, 256, 0, stream>>>(Xh, W1h, hraw, 64);
  stats_kernel<<<64, 256, 0, stream>>>(hraw, s0, s0 + 64);
  finalize_bn<<<1, 64, 0, stream>>>(s0, s0 + 64, bn1_g, bn1_b, s0 + 128, s0 + 192);
  bn_relu_to_f16<<<2048, 256, 0, stream>>>(hraw, s0 + 128, s0 + 192, nfh);

  // PQ[8192,512] = nfh @ [Fcomb | Fbot]
  // waves = (8192/32)*(512/32) = 4096 -> 512 blocks
  gemm_wmma<<<512, 256, 0, stream>>>(nfh, FcatT, PQ, 512);

  // edge mixing + max over K
  edge_kernel<<<MM, 256, 0, stream>>>(PQ, nn_idx, etype, aggraw);
  stats_kernel<<<64, 256, 0, stream>>>(aggraw, s1, s1 + 64);
  finalize_bn<<<1, 64, 0, stream>>>(s1, s1 + 64, bnm_g, bnm_b, s1 + 128, s1 + 192);
  bn_relu_to_f16<<<2048, 256, 0, stream>>>(aggraw, s1 + 128, s1 + 192, aggh);

  // conv2
  gemm_wmma<<<64, 256, 0, stream>>>(aggh, W2h, outraw, 64);
  stats_kernel<<<64, 256, 0, stream>>>(outraw, s2, s2 + 64);
  finalize_bn<<<1, 64, 0, stream>>>(s2, s2 + 64, bn2_g, bn2_b, s2 + 128, s2 + 192);
  final_kernel<<<2048, 256, 0, stream>>>(outraw, s2 + 128, s2 + 192, nf_in, out);
}
